// MultiScaleAttention_30958124269778
// MI455X (gfx1250) — compile-verified, matrix operand order checked by reading, not run on hardware
//
#include <hip/hip_runtime.h>
#include <hip/hip_bf16.h>
#include <math.h>
#include <stdint.h>

typedef __bf16 bf16;
typedef __attribute__((ext_vector_type(16))) __bf16    v16bf;
typedef __attribute__((ext_vector_type(8)))  float     v8f;
typedef __attribute__((ext_vector_type(4)))  uint32_t  v4u;
typedef __attribute__((ext_vector_type(8)))  uint32_t  v8u;

#define WMMA_BF16(a, b, c) \
  __builtin_amdgcn_wmma_f32_16x16x32_bf16(false, (a), false, (b), (short)0, (c), false, false)

// Low 32 bits of a generic (flat) pointer into LDS == workgroup-relative LDS
// byte address (ISA 10.2 aperture mapping).
__device__ __forceinline__ uint32_t lds_addr32(const void* p) {
  return (uint32_t)(uintptr_t)p;
}

// ---------------------------------------------------------------------------
// TDM: issue a 2D tile load (rows x elems_per_row bf16, global row stride in
// elements) into contiguous LDS. One instruction per wave; tracked by
// TENSORcnt. D# layout per CDNA5 ISA.
// ---------------------------------------------------------------------------
__device__ __forceinline__ void tdm_load_tile_bf16(
    uint32_t lds_byte_addr, uint64_t gaddr,
    uint32_t elems_per_row, uint32_t rows,
    uint32_t row_stride_elems, uint32_t tdim0, uint32_t tdim1) {
  v4u g0;
  g0[0] = 1u;                                   // count=1, user descriptor
  g0[1] = lds_byte_addr;                        // lds_addr
  g0[2] = (uint32_t)gaddr;                      // global_addr[31:0]
  g0[3] = (uint32_t)(gaddr >> 32) | (2u << 30); // global_addr[56:32] | type=2
  v8u g1;
  g1[0] = (1u << 16);                           // data_size=1 (2 bytes)
  g1[1] = (tdim0 & 0xFFFFu) << 16;              // tensor_dim0[15:0]
  g1[2] = (tdim0 >> 16) | ((tdim1 & 0xFFFFu) << 16);
  g1[3] = (tdim1 >> 16) | ((elems_per_row & 0xFFFFu) << 16);  // tile_dim0
  g1[4] = rows & 0xFFFFu;                       // tile_dim1 (tile_dim2=0)
  g1[5] = row_stride_elems;                     // tensor_dim0_stride[31:0]
  g1[6] = 0u;
  g1[7] = 0u;
  asm volatile("tensor_load_to_lds %0, %1" :: "s"(g0), "s"(g1) : "memory");
}

// Per-lane async global->LDS 16-byte copy (ASYNCcnt path).
__device__ __forceinline__ void async_copy_b128(uint32_t lds_byte_addr,
                                                const void* gptr) {
  asm volatile("global_load_async_to_lds_b128 %0, %1, off"
               :: "v"(lds_byte_addr), "v"(gptr) : "memory");
}
__device__ __forceinline__ void wait_async0() {
  asm volatile("s_wait_asynccnt 0x0" ::: "memory");
}

// ---------------------------------------------------------------------------
// Elementwise f32 -> bf16 cast
// ---------------------------------------------------------------------------
__global__ __launch_bounds__(256) void msa_cast_f32_bf16(const float* __restrict__ in,
                                                         bf16* __restrict__ out, int n) {
  int i = blockIdx.x * blockDim.x + threadIdx.x;
  if (i < n) out[i] = (bf16)in[i];
}

// ---------------------------------------------------------------------------
// WMMA GEMM with TDM double-buffered staging:
//   C[M,N] = A[M,K] * W[N,K]^T + bias[N]
// Block 256 threads (8 waves); block tile 128(M) x 64(N); wave tile 32x32.
// Wave 0 DMAs the A tile, wave 1 the B tile, for iteration k+1 while all
// waves run WMMAs on iteration k (TENSORcnt overlap).
// Requires 128|M, 64|N, 32|K (true for all shapes here).
// ---------------------------------------------------------------------------
__global__ __launch_bounds__(256) void msa_gemm_bf16(
    const bf16* __restrict__ A, const bf16* __restrict__ W,
    const float* __restrict__ bias,
    float* __restrict__ Cf, bf16* __restrict__ Cb,
    int M, int N, int K, int ldc, int col_off) {
  __shared__ __align__(64) bf16 ldsA[2][128 * 32];
  __shared__ __align__(64) bf16 ldsB[2][64 * 32];

  const int tid  = threadIdx.x;
  const int lane = tid & 31;
  const int wave = tid >> 5;          // 0..7
  const int wm   = wave >> 1;         // 0..3  (M direction)
  const int wn   = wave & 1;          // 0..1  (N direction)
  const int l15  = lane & 15;
  const int lhf  = lane >> 4;
  const int m0   = blockIdx.x * 128;
  const int n0   = blockIdx.y * 64;

  const uint64_t baseA = (uint64_t)(uintptr_t)A;
  const uint64_t baseW = (uint64_t)(uintptr_t)W;

  v8f acc[2][2] = {};

  // Prologue: DMA tile k0=0 into buffer 0.
  if (wave == 0)
    tdm_load_tile_bf16(lds_addr32(&ldsA[0][0]),
                       baseA + ((uint64_t)m0 * K) * 2, 32, 128, K, K, M - m0);
  if (wave == 1)
    tdm_load_tile_bf16(lds_addr32(&ldsB[0][0]),
                       baseW + ((uint64_t)n0 * K) * 2, 32, 64, K, K, N - n0);
  __builtin_amdgcn_s_wait_tensorcnt(0);
  __syncthreads();

  for (int k0 = 0; k0 < K; k0 += 32) {
    const int cur = (k0 >> 5) & 1;
    // Kick off the DMA for the next K tile into the other buffer.
    if (k0 + 32 < K) {
      const int nxt = cur ^ 1;
      if (wave == 0)
        tdm_load_tile_bf16(lds_addr32(&ldsA[nxt][0]),
                           baseA + ((uint64_t)m0 * K + k0 + 32) * 2,
                           32, 128, K, K - (k0 + 32), M - m0);
      if (wave == 1)
        tdm_load_tile_bf16(lds_addr32(&ldsB[nxt][0]),
                           baseW + ((uint64_t)n0 * K + k0 + 32) * 2,
                           32, 64, K, K - (k0 + 32), N - n0);
    }

    // Compute on the current buffer.
    v16bf af[2], bf[2];
#pragma unroll
    for (int i = 0; i < 2; i++)
      af[i] = *(const v16bf*)&ldsA[cur][(wm * 32 + i * 16 + l15) * 32 + lhf * 16];
#pragma unroll
    for (int j = 0; j < 2; j++)
      bf[j] = *(const v16bf*)&ldsB[cur][(wn * 32 + j * 16 + l15) * 32 + lhf * 16];
#pragma unroll
    for (int i = 0; i < 2; i++)
#pragma unroll
      for (int j = 0; j < 2; j++)
        acc[i][j] = WMMA_BF16(af[i], bf[j], acc[i][j]);

    __builtin_amdgcn_s_wait_tensorcnt(0);   // issuing waves drain their DMA
    __syncthreads();                        // everyone sees the next buffer
  }

  // Write back (C layout: VGPR e -> row e + 8*lhf, col l15)
#pragma unroll
  for (int i = 0; i < 2; i++)
#pragma unroll
    for (int j = 0; j < 2; j++) {
      int colb = n0 + wn * 32 + j * 16 + l15;
      float bv = bias ? bias[colb] : 0.0f;
#pragma unroll
      for (int e = 0; e < 8; e++) {
        int row = m0 + wm * 32 + i * 16 + e + lhf * 8;
        float v = acc[i][j][e] + bv;
        size_t idx = (size_t)row * ldc + col_off + colb;
        if (Cb) Cb[idx] = (bf16)v;
        else    Cf[idx] = v;
      }
    }
}

// ---------------------------------------------------------------------------
// Fused flash attention for one branch.
// qkv: [B*S, 3072] bf16, layout [Q | K | V] each 1024 cols, head-major.
// ctx: [B*S, 1024] bf16 context output.
// Band mask: key j visible to query i iff  j >= i - window.
// Block = 128 threads (4 waves); block covers 64 query rows of one (b,h);
// each wave owns one 16-row tile; streams keys in tiles of 32.
// K tile staged via async global->LDS; V tile transposed via scalar gather.
// For the long branch (mbuf/lbuf != null) also stores per-row softmax stats.
// ---------------------------------------------------------------------------
template <int DK>
__global__ __launch_bounds__(128) void msa_flash(
    const bf16* __restrict__ qkv, bf16* __restrict__ ctx,
    float* __restrict__ mbuf, float* __restrict__ lbuf,
    int H, int S, int window) {
  constexpr int D3  = 3072;
  constexpr int NKC = DK / 32;   // QK k-chunks
  constexpr int NO  = DK / 16;   // output column groups
  __shared__ __align__(64) bf16 ldsK[DK * 32];       // B-frag-ready K tile
  __shared__ __align__(64) bf16 ldsV[DK * 32];       // B-frag-ready V tile
  __shared__ __align__(64) bf16 ldsP[4][16 * 32];    // per-wave P tile

  const int tid  = threadIdx.x;
  const int lane = tid & 31;
  const int wave = tid >> 5;
  const int l15  = lane & 15;
  const int lhf  = lane >> 4;

  const int qblocks = S / 64;
  int bid = blockIdx.x;
  int qb  = bid % qblocks;
  int t   = bid / qblocks;
  int h   = t % H;
  int b   = t / H;
  const int q0b = qb * 64;
  const int q0  = q0b + wave * 16;
  const float sc = rsqrtf((float)DK);
  const int qcol = h * DK;
  const int kcol = 1024 + h * DK;
  const int vcol = 2048 + h * DK;

  // Q fragments (A layout): lane row = q0 + l15, contiguous 16 bf16 per chunk
  v16bf qf[NKC];
#pragma unroll
  for (int c = 0; c < NKC; c++)
    qf[c] = *(const v16bf*)&qkv[(size_t)(b * S + q0 + l15) * D3 + qcol + c * 32 + lhf * 16];

  v8f acc[NO] = {};
  float mrow[8], lrow[8];
#pragma unroll
  for (int e = 0; e < 8; e++) { mrow[e] = -1.0e30f; lrow[e] = 0.0f; }

  for (int kv0 = 0; kv0 < S; kv0 += 32) {
    // Skip key tiles fully in the masked past for ALL rows of this block
    if (kv0 + 31 < q0b - window) continue;   // block-uniform
    __syncthreads();

    // Stage K tile (32 keys x DK) in B-frag-ready layout with async copies:
    // ldsK[(g*32+kv)*16 + t] = K[kv0+kv][g*16+t]; 16-byte lane chunks.
    for (int c = tid; c < DK * 4; c += 128) {
      int half = c & 1, kv = (c >> 1) & 31, g = c >> 6;
      async_copy_b128(
          lds_addr32(&ldsK[((g << 5) + kv) * 16 + (half << 3)]),
          &qkv[(size_t)(b * S + kv0 + kv) * D3 + kcol + g * 16 + half * 8]);
    }
    // Stage V tile transposed (B-frag-ready for PV):
    // ldsV[((o*2+kh)*16+n)*16 + t] = V[kv0+kh*16+t][o*16+n]
    for (int c = tid; c < DK * 2; c += 128) {
      int o = c >> 5, kh = (c >> 4) & 1, n = c & 15;
      bf16 tv[16];
#pragma unroll
      for (int tt = 0; tt < 16; tt++)
        tv[tt] = qkv[(size_t)(b * S + kv0 + kh * 16 + tt) * D3 + vcol + o * 16 + n];
      bf16* dst = &ldsV[(((o * 2 + kh) * 16 + n)) << 4];
#pragma unroll
      for (int tt = 0; tt < 16; tt++) dst[tt] = tv[tt];
    }
    wait_async0();
    __syncthreads();

    // --- S = Q K^T  (two 16x16 score tiles) ---
    v8f st[2];
#pragma unroll
    for (int jt = 0; jt < 2; jt++) {
      v8f s = {};
#pragma unroll
      for (int c = 0; c < NKC; c++) {
        v16bf kb = *(const v16bf*)&ldsK[(((c * 2 + lhf) << 5) + jt * 16 + l15) << 4];
        s = WMMA_BF16(qf[c], kb, s);
      }
      st[jt] = s;
    }

    // --- mask + online softmax (rows live across the 16 lanes of a half) ---
    float rmax[8];
#pragma unroll
    for (int e = 0; e < 8; e++) rmax[e] = -1.0e30f;
#pragma unroll
    for (int jt = 0; jt < 2; jt++)
#pragma unroll
      for (int e = 0; e < 8; e++) {
        int i = q0 + e + lhf * 8;
        int j = kv0 + jt * 16 + l15;
        float v = st[jt][e] * sc + ((j >= i - window) ? 0.0f : -1.0e30f);
        st[jt][e] = v;
        rmax[e] = fmaxf(rmax[e], v);
      }
    float scl[8];
#pragma unroll
    for (int e = 0; e < 8; e++) {
      float v = rmax[e];
#pragma unroll
      for (int off = 1; off < 16; off <<= 1) v = fmaxf(v, __shfl_xor(v, off, 32));
      float mn = fmaxf(mrow[e], v);
      scl[e]  = __expf(mrow[e] - mn);
      mrow[e] = mn;
    }
    float rsum[8];
#pragma unroll
    for (int e = 0; e < 8; e++) rsum[e] = 0.0f;
#pragma unroll
    for (int jt = 0; jt < 2; jt++)
#pragma unroll
      for (int e = 0; e < 8; e++) {
        float p = __expf(st[jt][e] - mrow[e]);
        rsum[e] += p;
        ldsP[wave][(e + lhf * 8) * 32 + jt * 16 + l15] = (bf16)p;
      }
#pragma unroll
    for (int e = 0; e < 8; e++) {
      float v = rsum[e];
#pragma unroll
      for (int off = 1; off < 16; off <<= 1) v += __shfl_xor(v, off, 32);
      lrow[e] = lrow[e] * scl[e] + v;
    }
#pragma unroll
    for (int o = 0; o < NO; o++)
#pragma unroll
      for (int e = 0; e < 8; e++) acc[o][e] = acc[o][e] * scl[e];

    __syncthreads();   // make P visible (uniform)

    // --- O += P V  (A frag = P 16x32 from LDS, B frag = V) ---
    v16bf pf = *(const v16bf*)&ldsP[wave][l15 * 32 + lhf * 16];
#pragma unroll
    for (int o = 0; o < NO; o++) {
      v16bf vf = *(const v16bf*)&ldsV[(((o * 2 + lhf) * 16 + l15)) << 4];
      acc[o] = WMMA_BF16(pf, vf, acc[o]);
    }
  }

  // Epilogue: O /= l, write context (head-major columns = torch concat layout)
#pragma unroll
  for (int o = 0; o < NO; o++)
#pragma unroll
    for (int e = 0; e < 8; e++) {
      float ov = acc[o][e] / lrow[e];
      int row  = b * S + q0 + e + lhf * 8;
      ctx[(size_t)row * 1024 + h * DK + o * 16 + l15] = (bf16)ov;
    }
  if (mbuf != nullptr && l15 == 0) {
#pragma unroll
    for (int e = 0; e < 8; e++) {
      int r = (b * H + h) * S + q0 + e + lhf * 8;
      mbuf[r] = mrow[e];
      lbuf[r] = lrow[e];
    }
  }
}

// ---------------------------------------------------------------------------
// Long-branch head-averaged attention weights:
// attn[b,i,j] = (1/16) * sum_h exp(q_h(i)·k_h(j)/8 - m[b,h,i]) / l[b,h,i]
// Block = 128 threads (4 waves) per (b, 16-row i-tile, 128-col j-chunk);
// loops heads, async-staging each head's 128x64 K chunk into LDS.
// Deterministic register accumulation (no atomics).
// ---------------------------------------------------------------------------
__global__ __launch_bounds__(128) void msa_attnw(
    const bf16* __restrict__ qkv, const float* __restrict__ mbuf,
    const float* __restrict__ lbuf, float* __restrict__ attn, int S) {
  constexpr int DK = 64, H = 16, D3 = 3072;
  __shared__ __align__(64) bf16 ldsK[4 * 128 * 16];   // (g,kv) chunks of 16

  const int tid  = threadIdx.x;
  const int lane = tid & 31;
  const int wave = tid >> 5;
  const int l15  = lane & 15;
  const int lhf  = lane >> 4;

  const int jchunks = S / 128;
  int bid = blockIdx.x;
  int jc  = bid % jchunks;
  int t   = bid / jchunks;
  int it  = t % (S / 16);
  int b   = t / (S / 16);
  const int i0 = it * 16;
  const int j0 = jc * 128;
  const float sc = 0.125f;   // 1/sqrt(64)

  v8f accw[2] = {};

  for (int h = 0; h < H; h++) {
    __syncthreads();
    // Stage K_h rows j0..j0+127, B-frag-ready, 16-byte async lane chunks:
    // ldsK[(g*128+kv)*16 + t] = K[j0+kv][g*16+t]
    for (int c = tid; c < 1024; c += 128) {
      int half = c & 1, kv = (c >> 1) & 127, g = c >> 8;
      async_copy_b128(
          lds_addr32(&ldsK[((g << 7) + kv) * 16 + (half << 3)]),
          &qkv[(size_t)(b * S + j0 + kv) * D3 + 1024 + h * DK + g * 16 + half * 8]);
    }
    wait_async0();
    __syncthreads();

    v16bf qf[2];
#pragma unroll
    for (int c = 0; c < 2; c++)
      qf[c] = *(const v16bf*)&qkv[(size_t)(b * S + i0 + l15) * D3 + h * DK + c * 32 + lhf * 16];

    float mr[8], lr[8];
#pragma unroll
    for (int e = 0; e < 8; e++) {
      int r = (b * H + h) * S + i0 + e + lhf * 8;
      mr[e] = mbuf[r];
      lr[e] = lbuf[r];
    }
#pragma unroll
    for (int jt = 0; jt < 2; jt++) {
      v8f s = {};
#pragma unroll
      for (int c = 0; c < 2; c++) {
        int kv = wave * 32 + jt * 16 + l15;
        v16bf kb = *(const v16bf*)&ldsK[(((c * 2 + lhf) << 7) + kv) << 4];
        s = WMMA_BF16(qf[c], kb, s);
      }
#pragma unroll
      for (int e = 0; e < 8; e++)
        accw[jt][e] += __expf(s[e] * sc - mr[e]) / (lr[e] * 16.0f);
    }
  }
#pragma unroll
  for (int jt = 0; jt < 2; jt++)
#pragma unroll
    for (int e = 0; e < 8; e++) {
      int i = i0 + e + lhf * 8;
      int j = j0 + wave * 32 + jt * 16 + l15;
      attn[(size_t)(b * S + i) * S + j] = accw[jt][e];
    }
}

// ---------------------------------------------------------------------------
// Residual + LayerNorm:  out = LN(comb + x) * gamma + beta,  D = 1024
// ---------------------------------------------------------------------------
__global__ __launch_bounds__(256) void msa_ln(
    const float* __restrict__ comb, const float* __restrict__ x,
    const float* __restrict__ gamma, const float* __restrict__ beta,
    float* __restrict__ out, int D) {
  __shared__ float red[16];
  const int row = blockIdx.x;
  const int tid = threadIdx.x;
  const float* c  = comb + (size_t)row * D;
  const float* xr = x + (size_t)row * D;
  float s = 0.f, s2 = 0.f, vals[4];
  int q = 0;
  for (int idx = tid; idx < D; idx += 256, q++) {
    float v = c[idx] + xr[idx];
    vals[q] = v;
    s += v;
    s2 += v * v;
  }
  for (int off = 16; off > 0; off >>= 1) {
    s  += __shfl_down(s, off, 32);
    s2 += __shfl_down(s2, off, 32);
  }
  int wave = tid >> 5, lane = tid & 31;
  if (lane == 0) { red[wave] = s; red[8 + wave] = s2; }
  __syncthreads();
  if (tid == 0) {
    float a = 0.f, b2 = 0.f;
    for (int w = 0; w < 8; w++) { a += red[w]; b2 += red[8 + w]; }
    red[0] = a; red[8] = b2;
  }
  __syncthreads();
  float mu  = red[0] / D;
  float var = red[8] / D - mu * mu;
  float inv = rsqrtf(var + 1e-5f);
  q = 0;
  for (int idx = tid; idx < D; idx += 256, q++)
    out[(size_t)row * D + idx] = (vals[q] - mu) * inv * gamma[idx] + beta[idx];
}

// ---------------------------------------------------------------------------
extern "C" void kernel_launch(void* const* d_in, const int* in_sizes, int n_in,
                              void* d_out, int out_size, void* d_ws, size_t ws_size,
                              hipStream_t stream) {
  (void)in_sizes; (void)n_in; (void)out_size; (void)ws_size;
  const int B = 2, S = 2048, D = 1024, BS = B * S, D3 = 3 * D, H = 16;

  const float* x        = (const float*)d_in[0];
  const float* w_in[3]  = {(const float*)d_in[1], (const float*)d_in[5], (const float*)d_in[9]};
  const float* b_in[3]  = {(const float*)d_in[2], (const float*)d_in[6], (const float*)d_in[10]};
  const float* w_o[3]   = {(const float*)d_in[3], (const float*)d_in[7], (const float*)d_in[11]};
  const float* b_o[3]   = {(const float*)d_in[4], (const float*)d_in[8], (const float*)d_in[12]};
  const float* w_comb   = (const float*)d_in[13];
  const float* b_comb   = (const float*)d_in[14];
  const float* gamma    = (const float*)d_in[15];
  const float* beta     = (const float*)d_in[16];

  // Workspace carve-out (~110 MB), 256B aligned slices
  char* ws = (char*)d_ws;
  size_t off = 0;
  auto carve = [&](size_t bytes) -> char* {
    char* p = ws + off;
    off += (bytes + 255) & ~(size_t)255;
    return p;
  };
  bf16* xb  = (bf16*)carve((size_t)BS * D * 2);
  bf16* wqb[3]; for (int i = 0; i < 3; i++) wqb[i] = (bf16*)carve((size_t)D3 * D * 2);
  bf16* wob[3]; for (int i = 0; i < 3; i++) wob[i] = (bf16*)carve((size_t)D * D * 2);
  bf16* wcb   = (bf16*)carve((size_t)D * D3 * 2);
  bf16* qkv   = (bf16*)carve((size_t)BS * D3 * 2);   // reused per branch
  bf16* ctx   = (bf16*)carve((size_t)BS * D * 2);    // reused per branch
  bf16* cat   = (bf16*)carve((size_t)BS * D3 * 2);   // concat(short,med,long)
  float* comb = (float*)carve((size_t)BS * D * 4);
  float* mbuf = (float*)carve((size_t)B * H * S * 4);
  float* lbuf = (float*)carve((size_t)B * H * S * 4);

  auto cast = [&](const float* src, bf16* dst, size_t n) {
    msa_cast_f32_bf16<<<(int)((n + 255) / 256), 256, 0, stream>>>(src, dst, (int)n);
  };
  cast(x, xb, (size_t)BS * D);
  for (int i = 0; i < 3; i++) {
    cast(w_in[i], wqb[i], (size_t)D3 * D);
    cast(w_o[i],  wob[i], (size_t)D * D);
  }
  cast(w_comb, wcb, (size_t)D * D3);

  const int Hs[3]  = {8, 8, 16};
  const int win[3] = {10, 30, 1 << 28};

  for (int br = 0; br < 3; br++) {
    // QKV projection: [BS,D] x [D3,D]^T -> bf16 [BS,D3]
    dim3 g1(BS / 128, D3 / 64);
    msa_gemm_bf16<<<g1, 256, 0, stream>>>(xb, wqb[br], b_in[br], nullptr, qkv,
                                          BS, D3, D, D3, 0);
    // Fused attention
    int grid = B * Hs[br] * (S / 64);
    if (br < 2)
      msa_flash<128><<<grid, 128, 0, stream>>>(qkv, ctx, nullptr, nullptr,
                                               Hs[br], S, win[br]);
    else
      msa_flash<64><<<grid, 128, 0, stream>>>(qkv, ctx, mbuf, lbuf,
                                              Hs[br], S, win[br]);
    // Output projection into concat buffer columns [br*D, br*D+D)
    dim3 g2(BS / 128, D / 64);
    msa_gemm_bf16<<<g2, 256, 0, stream>>>(ctx, wob[br], b_o[br], nullptr, cat,
                                          BS, D, D, D3, br * D);
  }

  // Combine: [BS,3D] x [D,3D]^T -> f32 [BS,D]
  dim3 g3(BS / 128, D / 64);
  msa_gemm_bf16<<<g3, 256, 0, stream>>>(cat, wcb, b_comb, comb, nullptr,
                                        BS, D, D3, D, 0);

  float* out_ln   = (float*)d_out;
  float* out_attn = out_ln + (size_t)BS * D;

  msa_ln<<<BS, 256, 0, stream>>>(comb, x, gamma, beta, out_ln, D);
  msa_attnw<<<B * (S / 16) * (S / 128), 128, 0, stream>>>(qkv, mbuf, lbuf, out_attn, S);
}
